// LSTM_72739566125073
// MI455X (gfx1250) — matrix operand order repaired and checked by reading, not compiled, not run on hardware
//
#include <hip/hip_runtime.h>

typedef _Float16 f16;
typedef __attribute__((ext_vector_type(16))) _Float16 v16h;
typedef __attribute__((ext_vector_type(8)))  _Float16 v8h;
typedef __attribute__((ext_vector_type(8)))  float    v8f;
typedef __attribute__((ext_vector_type(4)))  float    v4f;

#define B_SZ   256
#define T_SZ   1024
#define HID    128
#define OUT_SZ 32

// d_out layout: out [B,T,32] | hN [2,B,128] | cN [2,B,128]
#define HN_OFF ((size_t)B_SZ * T_SZ * OUT_SZ)
#define CN_OFF (HN_OFF + (size_t)2 * B_SZ * HID)

union ABFrag { v16h v; v8h h[2]; };

__device__ __forceinline__ float sigf(float x) {
    return 1.0f / (1.0f + __expf(-x));
}
__device__ __forceinline__ float tanhfast(float x) {
    float e = __expf(-2.0f * x);
    return (1.0f - e) / (1.0f + e);
}

// CDNA5 async global->LDS copy (ASYNCcnt-tracked). lds_off is the byte offset
// within this workgroup's LDS allocation.
__device__ __forceinline__ void async_load_b128(unsigned lds_off, const void* gsrc) {
    asm volatile("global_load_async_to_lds_b128 %0, %1, off"
                 :: "v"(lds_off), "v"(gsrc) : "memory");
}
__device__ __forceinline__ void wait_asynccnt0() {
    asm volatile("s_wait_asynccnt 0" ::: "memory");
}

// x f32 -> f16 pre-conversion (one pass; makes both layers' x streams f16 so
// the per-step staging is a pure async byte copy).
__global__ __launch_bounds__(256) void cvt_x_kernel(
    const float* __restrict__ src, f16* __restrict__ dst, int n4) {
    int i = blockIdx.x * blockDim.x + threadIdx.x;
    if (i < n4) {
        v4f v = *(const v4f*)(src + (size_t)i * 4);
        f16* d = dst + (size_t)i * 4;
        d[0] = (f16)v[0]; d[1] = (f16)v[1];
        d[2] = (f16)v[2]; d[3] = (f16)v[3];
    }
}

// One LSTM layer, fused input+recurrent GEMM, persistent over all T.
// Grid: 16 workgroups (batch tiles of 16). Block: 256 threads = 8 waves.
// Wave w owns gate columns [w*16, w*16+16) of each gate i,f,g,o.
// x tile double-buffered in LDS via async loads; recurrent (h-part) weight
// fragments are hoisted into VGPRs (loop-invariant, 128 VGPRs).
template <int IN_F, bool HAS_FC>
__global__ __launch_bounds__(256) void lstm_layer_kernel(
    const f16*   __restrict__ xin,       // [B,T,IN_F] f16
    const float* __restrict__ Wih,       // [512, IN_F]
    const float* __restrict__ Whh,       // [512, 128]
    const float* __restrict__ bih,       // [512]
    const float* __restrict__ bhh,       // [512]
    const float* __restrict__ h0,        // [2,B,128]
    const float* __restrict__ c0,        // [2,B,128]
    int layer,
    f16*         __restrict__ out_h_f16, // !HAS_FC: h stream [B,T,128] f16
    const float* __restrict__ fc_w,      // [32,128] (HAS_FC)
    const float* __restrict__ fc_b,      // [32]     (HAS_FC)
    float*       __restrict__ d_out)
{
    constexpr int KTOT = IN_F + HID;        // combined K: [x | h]
    constexpr int KC   = KTOT / 32;         // WMMA K-chunks of 32
    constexpr int XC   = IN_F / 32;         // chunks in the x part
    constexpr int HC   = HID / 32;          // chunks in the h part (4)
    constexpr int CH   = IN_F / 8;          // 16B chunks per x row

    // LDS carve (dynamic allocation starts at LDS offset 0: no static shared)
    constexpr unsigned W_OFF  = 0;
    constexpr unsigned HB_OFF = W_OFF + 512u * KTOT * 2;
    constexpr unsigned XB_OFF = HB_OFF + 16u * HID * 2;
    constexpr unsigned XB_SZ  = 16u * IN_F * 2;
    constexpr unsigned FC_OFF = XB_OFF + 2u * XB_SZ;

    extern __shared__ char smem_raw[];
    f16* w_lds  = (f16*)(smem_raw + W_OFF);    // [512][KTOT] combined weights
    f16* hbuf   = (f16*)(smem_raw + HB_OFF);   // [16][128]   h state
    f16* xbuf   = (f16*)(smem_raw + XB_OFF);   // [2][16][IN_F] x double buffer
    f16* fc_lds = (f16*)(smem_raw + FC_OFF);   // [32][128]   fc weights

    const int tid   = threadIdx.x;
    const int wave  = tid >> 5;
    const int lane  = tid & 31;
    const int lhalf = lane >> 4;    // 0: lanes 0-15, 1: lanes 16-31
    const int lcol  = lane & 15;
    const int b0    = blockIdx.x * 16;

    // ---- stage combined [W_ih ; W_hh] -> LDS as f16, [col][K] ----
    for (int idx = tid; idx < 512 * KTOT; idx += 256) {
        int col = idx / KTOT, k = idx % KTOT;
        float v = (k < IN_F) ? Wih[col * IN_F + k] : Whh[col * HID + (k - IN_F)];
        w_lds[idx] = (f16)v;
    }
    if (HAS_FC) {
        for (int idx = tid; idx < OUT_SZ * HID; idx += 256)
            fc_lds[idx] = (f16)fc_w[idx];
    }

    // ---- per-lane state: c for 8 (row,col) pairs; bias; h0 -> LDS ----
    const int myj = wave * 16 + lcol;            // gate-internal column 0..127
    float bias[4];
#pragma unroll
    for (int g = 0; g < 4; ++g)
        bias[g] = bih[g * HID + myj] + bhh[g * HID + myj];

    float c_reg[8];
#pragma unroll
    for (int r = 0; r < 8; ++r) {
        int row = r + 8 * lhalf;
        size_t st = (size_t)layer * B_SZ * HID + (size_t)(b0 + row) * HID + myj;
        c_reg[r] = c0[st];
        hbuf[row * HID + myj] = (f16)h0[st];
    }

    // ---- prefetch x_0 into xbuf[0] (one async b128 per participating thread)
    if (tid < 16 * CH) {
        int row = tid / CH, ch = tid % CH;
        async_load_b128(XB_OFF + (unsigned)(row * IN_F + ch * 8) * 2,
                        xin + ((size_t)(b0 + row) * T_SZ + 0) * IN_F + ch * 8);
    }

    __syncthreads();   // weight image complete -> hoist recurrent B frags

    // ---- hoist h-part (recurrent) weight fragments into registers ----
    // bh[kh][g]: K-chunk (XC+kh), gate g, columns of this wave. 128 VGPRs.
    ABFrag bh[HC][4];
#pragma unroll
    for (int kh = 0; kh < HC; ++kh) {
#pragma unroll
        for (int g = 0; g < 4; ++g) {
            const f16* brow = w_lds + (size_t)(g * HID + myj) * KTOT +
                              (XC + kh) * 32 + lhalf * 16;
            bh[kh][g].h[0] = *(const v8h*)(brow);
            bh[kh][g].h[1] = *(const v8h*)(brow + 8);
        }
    }

    for (int t = 0; t < T_SZ; ++t) {
        const int buf = t & 1;

        wait_asynccnt0();      // this wave's async loads for xbuf[buf] done
        __syncthreads();       // barrier A: x_t and h(t-1) visible

        // ---- prefetch x_{t+1} into the other buffer while GEMM runs ----
        if (t + 1 < T_SZ) {
            if (tid < 16 * CH) {
                int row = tid / CH, ch = tid % CH;
                async_load_b128(
                    XB_OFF + (buf ^ 1) * XB_SZ +
                        (unsigned)(row * IN_F + ch * 8) * 2,
                    xin + ((size_t)(b0 + row) * T_SZ + (t + 1)) * IN_F + ch * 8);
            }
        }

        // ---- gates[16 x 16-per-gate] = [x|h] x Wc, f32 accumulate ----
        v8f acc[4] = {};       // zero C -> inline 0 operand on first WMMA

        // x-part: B frags from LDS weight image
#pragma unroll
        for (int kc = 0; kc < XC; ++kc) {
            const f16* arow = xbuf + buf * (16 * IN_F) + lcol * IN_F + kc * 32;
            ABFrag a;
            a.h[0] = *(const v8h*)(arow + lhalf * 16);
            a.h[1] = *(const v8h*)(arow + lhalf * 16 + 8);
#pragma unroll
            for (int g = 0; g < 4; ++g) {
                ABFrag b;
                const f16* brow = w_lds + (size_t)(g * HID + myj) * KTOT +
                                  kc * 32 + lhalf * 16;
                b.h[0] = *(const v8h*)(brow);
                b.h[1] = *(const v8h*)(brow + 8);
                acc[g] = __builtin_amdgcn_wmma_f32_16x16x32_f16(
                    false, a.v, false, b.v, (short)0, acc[g], false, false);
            }
        }
        // h-part: B frags already in registers; only A loads touch LDS
#pragma unroll
        for (int kh = 0; kh < HC; ++kh) {
            const f16* arow = hbuf + lcol * HID + kh * 32;
            ABFrag a;
            a.h[0] = *(const v8h*)(arow + lhalf * 16);
            a.h[1] = *(const v8h*)(arow + lhalf * 16 + 8);
#pragma unroll
            for (int g = 0; g < 4; ++g) {
                acc[g] = __builtin_amdgcn_wmma_f32_16x16x32_f16(
                    false, a.v, false, bh[kh][g].v, (short)0, acc[g],
                    false, false);
            }
        }

        // ---- bias + nonlinearity + state update (pure register math) ----
        float hval[8];
#pragma unroll
        for (int r = 0; r < 8; ++r) {
            float ig = sigf(acc[0][r] + bias[0]);
            float fg = sigf(acc[1][r] + bias[1]);
            float gg = tanhfast(acc[2][r] + bias[2]);
            float og = sigf(acc[3][r] + bias[3]);
            float c  = fg * c_reg[r] + ig * gg;
            c_reg[r] = c;
            hval[r]  = og * tanhfast(c);
        }
        __syncthreads();       // barrier B: all GEMM reads of x/h done

        // ---- publish h(t) ----
#pragma unroll
        for (int r = 0; r < 8; ++r) {
            int row = r + 8 * lhalf;
            hbuf[row * HID + myj] = (f16)hval[r];
        }
        if (!HAS_FC) {
            // stream h to workspace for the next layer (fire-and-forget)
#pragma unroll
            for (int r = 0; r < 8; ++r) {
                int row = r + 8 * lhalf;
                out_h_f16[((size_t)(b0 + row) * T_SZ + t) * HID + myj] =
                    (f16)hval[r];
            }
        } else {
            __syncthreads();   // barrier C: full h(t) visible in LDS
            // fused FC: out[16 x 32] = h x fc_w^T + fc_b ; waves 0,1 only
            if (__builtin_amdgcn_readfirstlane(wave) < 2) {
                int ocol = wave * 16 + lcol;
                v8f fa = {};
#pragma unroll
                for (int kc = 0; kc < HID / 32; ++kc) {
                    const f16* arow = hbuf + lcol * HID + kc * 32 + lhalf * 16;
                    ABFrag a;
                    a.h[0] = *(const v8h*)(arow);
                    a.h[1] = *(const v8h*)(arow + 8);
                    ABFrag b;
                    const f16* brow =
                        fc_lds + (size_t)ocol * HID + kc * 32 + lhalf * 16;
                    b.h[0] = *(const v8h*)(brow);
                    b.h[1] = *(const v8h*)(brow + 8);
                    fa = __builtin_amdgcn_wmma_f32_16x16x32_f16(
                        false, a.v, false, b.v, (short)0, fa, false, false);
                }
                float fb = fc_b[ocol];
#pragma unroll
                for (int r = 0; r < 8; ++r) {
                    int row = r + 8 * lhalf;
                    d_out[((size_t)(b0 + row) * T_SZ + t) * OUT_SZ + ocol] =
                        fa[r] + fb;
                }
            }
        }

        // ---- final states ----
        if (t == T_SZ - 1) {
#pragma unroll
            for (int r = 0; r < 8; ++r) {
                int row = r + 8 * lhalf;
                size_t st = (size_t)layer * B_SZ * HID +
                            (size_t)(b0 + row) * HID + myj;
                d_out[HN_OFF + st] = hval[r];
                d_out[CN_OFF + st] = c_reg[r];
            }
        }
    }
}

extern "C" void kernel_launch(void* const* d_in, const int* in_sizes, int n_in,
                              void* d_out, int out_size, void* d_ws, size_t ws_size,
                              hipStream_t stream) {
    (void)in_sizes; (void)n_in; (void)out_size; (void)ws_size;
    const float* x    = (const float*)d_in[0];
    const float* h0   = (const float*)d_in[1];
    const float* c0   = (const float*)d_in[2];
    const float* Wih0 = (const float*)d_in[3];
    const float* Whh0 = (const float*)d_in[4];
    const float* bih0 = (const float*)d_in[5];
    const float* bhh0 = (const float*)d_in[6];
    const float* Wih1 = (const float*)d_in[7];
    const float* Whh1 = (const float*)d_in[8];
    const float* bih1 = (const float*)d_in[9];
    const float* bhh1 = (const float*)d_in[10];
    const float* fcw  = (const float*)d_in[11];
    const float* fcb  = (const float*)d_in[12];
    float* out = (float*)d_out;

    // workspace: out0 f16 [B,T,128] (64MB) | x16 f16 [B,T,64] (32MB)
    f16* out0 = (f16*)d_ws;
    f16* x16  = (f16*)((char*)d_ws + (size_t)B_SZ * T_SZ * HID * sizeof(f16));

    constexpr size_t lds0 =
        (size_t)(512 * 192 + 16 * HID + 2 * 16 * 64) * sizeof(f16);
    constexpr size_t lds1 =
        (size_t)(512 * 256 + 16 * HID + 2 * 16 * 128 + 32 * 128) * sizeof(f16);

    hipFuncSetAttribute(
        reinterpret_cast<const void*>(lstm_layer_kernel<64, false>),
        hipFuncAttributeMaxDynamicSharedMemorySize, (int)lds0);
    hipFuncSetAttribute(
        reinterpret_cast<const void*>(lstm_layer_kernel<128, true>),
        hipFuncAttributeMaxDynamicSharedMemorySize, (int)lds1);

    // x f32 -> f16 (B*T*64 = 16.8M elements, 4 per thread)
    {
        int n4 = B_SZ * T_SZ * 64 / 4;
        cvt_x_kernel<<<(n4 + 255) / 256, 256, 0, stream>>>(x, x16, n4);
    }
    lstm_layer_kernel<64, false><<<16, 256, lds0, stream>>>(
        x16, Wih0, Whh0, bih0, bhh0, h0, c0, 0,
        out0, nullptr, nullptr, out);
    lstm_layer_kernel<128, true><<<16, 256, lds1, stream>>>(
        out0, Wih1, Whh1, bih1, bhh1, h0, c0, 1,
        nullptr, fcw, fcb, out);
}
